// PatternConsistencyLoss_68324339745174
// MI455X (gfx1250) — matrix operand order repaired and checked by reading, not compiled, not run on hardware
//
#include <hip/hip_runtime.h>
#include <hip/hip_bf16.h>

typedef __attribute__((ext_vector_type(16))) _Float16 v16h;
typedef __attribute__((ext_vector_type(8)))  float    v8f;

#define EPSF 1e-8f
#define Bn   8
#define Cn   3
#define Hn   192
#define Wn   192
#define PADn 48
#define PWn  288          // 48 + 192 + 48
#define NIMG 24           // images per input (B*C)
#define NLAG 97
#define NPIX (Hn * Wn)    // 36864
#define NBIN 127          // dy in [-63,63] -> bin = dy + 63

// ---- workspace layout (bytes) ----
static constexpr size_t OFF_PIMG = 0;            // 48 * 192*288*2 = 5,308,416
static constexpr size_t OFF_NORM = 5308416;      // 48 f32
static constexpr size_t OFF_CORR = 5308672;      // 2*24*97*97*4 = 1,806,528
static constexpr size_t OFF_SCAL = 7115200;      // [pmaxBits, tmaxBits, sumF32]
static constexpr long   NTOT     = (long)Bn * Bn * NLAG * NLAG;   // 602176

// ------------------------------------------------------------------
// K0: zero accumulators
__global__ void k_init(unsigned* scal) {
    if (threadIdx.x < 3) scal[threadIdx.x] = 0u;   // 0 bits == 0.0f
}

// ------------------------------------------------------------------
// K1: per-image normalize, store zero-padded f16 image, store ||xn||
__global__ __launch_bounds__(256)
void k_normalize(const float* __restrict__ pred,
                 const float* __restrict__ targ,
                 _Float16* __restrict__ pimgs,
                 float* __restrict__ norms) {
    const int img = blockIdx.x;            // 0..47
    const int s   = img / NIMG;
    const int r   = img % NIMG;            // b*3 + c
    const float* src = (s ? targ : pred) + (size_t)r * NPIX;

    __shared__ float red[256];
    float sum = 0.f, sq = 0.f;
    for (int i = threadIdx.x; i < NPIX; i += 256) {
        float v = src[i]; sum += v; sq += v * v;
    }
    red[threadIdx.x] = sum; __syncthreads();
    for (int o = 128; o > 0; o >>= 1) {
        if (threadIdx.x < o) red[threadIdx.x] += red[threadIdx.x + o];
        __syncthreads();
    }
    const float tsum = red[0]; __syncthreads();
    red[threadIdx.x] = sq; __syncthreads();
    for (int o = 128; o > 0; o >>= 1) {
        if (threadIdx.x < o) red[threadIdx.x] += red[threadIdx.x + o];
        __syncthreads();
    }
    const float tsq = red[0];

    const float mean = tsum / (float)NPIX;
    const float var  = fmaxf((tsq - tsum * mean) / (float)(NPIX - 1), 0.f);
    const float stdv = sqrtf(var);
    const float inv  = 1.f / (stdv + EPSF);

    _Float16* P = pimgs + (size_t)img * Hn * PWn;
    for (int i = threadIdx.x; i < Hn * PWn; i += 256) {
        const int row = i / PWn, col = i % PWn;
        float v = 0.f;
        if (col >= PADn && col < PADn + Wn)
            v = (src[row * Wn + col - PADn] - mean) * inv;
        P[i] = (_Float16)v;
    }
    if (threadIdx.x == 0)
        norms[img] = sqrtf(var * (float)(NPIX - 1) * inv * inv);  // ||xn||_2
}

// ------------------------------------------------------------------
// K2: banded WMMA correlation. One block per (sdx, image, input).
//     corr[dy,sdx] = sum_h <X[h,:], X'[h+dy,:]> over band tiles |tj-ti|<=3.
//     Each wave owns a tile-column tj: B fragments (shifted image) loaded
//     once into registers for the whole column, A fragments per tile.
__global__ __launch_bounds__(256)
void k_corr_wmma(const _Float16* __restrict__ pimgs,
                 float* __restrict__ corr) {
    const int dxI = blockIdx.x;            // 0..96
    const int img = blockIdx.y;            // 0..23
    const int s   = blockIdx.z;            // 0..1
    const int sdx = dxI - PADn;            // [-48, 48]
    const _Float16* P = pimgs + (size_t)(s * NIMG + img) * Hn * PWn;

    __shared__ float sCorr[NBIN];
    if (threadIdx.x < NBIN) sCorr[threadIdx.x] = 0.f;
    __syncthreads();

    const int lane = threadIdx.x & 31;
    const int warp = threadIdx.x >> 5;     // 8 waves
    const int half = lane >> 4;
    const int n    = lane & 15;            // A row (m) and B col (n) lane index

    for (int tj = warp; tj < 12; tj += 8) {       // waves 0-3 take a 2nd column
        // ---- load all B fragments (column tj, full K=192) once ----
        const _Float16* brow =
            P + (size_t)(16 * tj + n) * PWn + PADn + sdx + 16 * half;
        v16h bf[6];
        #pragma unroll
        for (int kk = 0; kk < 6; ++kk)
            #pragma unroll
            for (int e = 0; e < 16; ++e) bf[kk][e] = brow[32 * kk + e];

        const int tlo = (tj > 3) ? tj - 3 : 0;
        const int thi = (tj < 8) ? tj + 3 : 11;
        for (int ti = tlo; ti <= thi; ++ti) {
            // ---- A fragments for tile row ti (one load clause) ----
            const _Float16* arow =
                P + (size_t)(16 * ti + n) * PWn + PADn + 16 * half;
            v16h af[6];
            #pragma unroll
            for (int kk = 0; kk < 6; ++kk) {
                #pragma unroll
                for (int e = 0; e < 8; ++e) af[kk][e]     = arow[32 * kk + e];
                #pragma unroll
                for (int e = 0; e < 8; ++e) af[kk][e + 8] = arow[32 * kk + 8 + e];
            }
            // ---- 6 chained WMMAs over K ----
            v8f acc = {};
            #pragma unroll
            for (int kk = 0; kk < 6; ++kk)
                acc = __builtin_amdgcn_wmma_f32_16x16x32_f16(
                          false, af[kk], false, bf[kk], (short)0, acc,
                          false, false);
            // ---- unconditional diagonal reduction into 127 bins ----
            const int base = 16 * (tj - ti) + n - 8 * half + 63;
            #pragma unroll
            for (int rr = 0; rr < 8; ++rr)
                atomicAdd(&sCorr[base - rr], acc[rr]);   // ds_add_f32
        }
    }
    __syncthreads();

    float* out = corr + (size_t)(s * NIMG + img) * NLAG * NLAG;   // [dy][dx]
    for (int dy = threadIdx.x; dy < NLAG; dy += 256)
        out[(size_t)dy * NLAG + dxI] = sCorr[dy + 15];   // bin = (dy-48)+63
}

// ------------------------------------------------------------------
// combined corr value: mean over channels of corr[b,c]/(norm[b2,c]+eps)
__device__ __forceinline__ float corr_comb(const float* __restrict__ corr,
                                           const float* __restrict__ norms,
                                           int s, int b2, int b, int i, int j) {
    float acc = 0.f;
    #pragma unroll
    for (int c = 0; c < Cn; ++c) {
        const float cv = corr[((size_t)(s * NIMG + b * Cn + c) * NLAG + i) * NLAG + j];
        const float nm = norms[s * NIMG + b2 * Cn + c];
        acc += cv / (nm + EPSF);
    }
    return acc * (1.f / 3.f);
}

// K3a: global max of combined pred / target tensors (positive peak -> bit max)
__global__ __launch_bounds__(256)
void k_max(const float* __restrict__ corr, const float* __restrict__ norms,
           unsigned* __restrict__ scal) {
    float pm = 0.f, tm = 0.f;
    const long stride = (long)gridDim.x * 256;
    for (long idx = (long)blockIdx.x * 256 + threadIdx.x; idx < NTOT; idx += stride) {
        long e = idx;
        const int j  = (int)(e % NLAG); e /= NLAG;
        const int i  = (int)(e % NLAG); e /= NLAG;
        const int b  = (int)(e % Bn);
        const int b2 = (int)(e / Bn);
        pm = fmaxf(pm, corr_comb(corr, norms, 0, b2, b, i, j));
        tm = fmaxf(tm, corr_comb(corr, norms, 1, b2, b, i, j));
    }
    __shared__ float red[256];
    red[threadIdx.x] = pm; __syncthreads();
    for (int o = 128; o > 0; o >>= 1) {
        if (threadIdx.x < o) red[threadIdx.x] = fmaxf(red[threadIdx.x], red[threadIdx.x + o]);
        __syncthreads();
    }
    if (threadIdx.x == 0) atomicMax(&scal[0], __float_as_uint(red[0]));
    __syncthreads();
    red[threadIdx.x] = tm; __syncthreads();
    for (int o = 128; o > 0; o >>= 1) {
        if (threadIdx.x < o) red[threadIdx.x] = fmaxf(red[threadIdx.x], red[threadIdx.x + o]);
        __syncthreads();
    }
    if (threadIdx.x == 0) atomicMax(&scal[1], __float_as_uint(red[0]));
}

// K3b: sum |p/(pmax+eps) - t/(tmax+eps)|
__global__ __launch_bounds__(256)
void k_loss(const float* __restrict__ corr, const float* __restrict__ norms,
            unsigned* __restrict__ scal) {
    const float ip = 1.f / (__uint_as_float(scal[0]) + EPSF);
    const float it = 1.f / (__uint_as_float(scal[1]) + EPSF);
    float acc = 0.f;
    const long stride = (long)gridDim.x * 256;
    for (long idx = (long)blockIdx.x * 256 + threadIdx.x; idx < NTOT; idx += stride) {
        long e = idx;
        const int j  = (int)(e % NLAG); e /= NLAG;
        const int i  = (int)(e % NLAG); e /= NLAG;
        const int b  = (int)(e % Bn);
        const int b2 = (int)(e / Bn);
        const float p = corr_comb(corr, norms, 0, b2, b, i, j) * ip;
        const float t = corr_comb(corr, norms, 1, b2, b, i, j) * it;
        acc += fabsf(p - t);
    }
    __shared__ float red[256];
    red[threadIdx.x] = acc; __syncthreads();
    for (int o = 128; o > 0; o >>= 1) {
        if (threadIdx.x < o) red[threadIdx.x] += red[threadIdx.x + o];
        __syncthreads();
    }
    if (threadIdx.x == 0) atomicAdd((float*)&scal[2], red[0]);
}

// K3c: final scalar
__global__ void k_final(const unsigned* __restrict__ scal, float* __restrict__ out) {
    out[0] = __uint_as_float(scal[2]) / (float)NTOT;
}

// ------------------------------------------------------------------
extern "C" void kernel_launch(void* const* d_in, const int* in_sizes, int n_in,
                              void* d_out, int out_size, void* d_ws, size_t ws_size,
                              hipStream_t stream) {
    const float* pred = (const float*)d_in[0];
    const float* targ = (const float*)d_in[1];
    float* out        = (float*)d_out;

    char* ws = (char*)d_ws;
    _Float16* pimgs = (_Float16*)(ws + OFF_PIMG);
    float*    norms = (float*)   (ws + OFF_NORM);
    float*    corr  = (float*)   (ws + OFF_CORR);
    unsigned* scal  = (unsigned*)(ws + OFF_SCAL);

    k_init<<<1, 32, 0, stream>>>(scal);
    k_normalize<<<48, 256, 0, stream>>>(pred, targ, pimgs, norms);
    k_corr_wmma<<<dim3(NLAG, NIMG, 2), 256, 0, stream>>>(pimgs, corr);
    k_max <<<1024, 256, 0, stream>>>(corr, norms, scal);
    k_loss<<<1024, 256, 0, stream>>>(corr, norms, scal);
    k_final<<<1, 1, 0, stream>>>(scal, out);
}
